// _MultiHeadSelfAttention_79130477462132
// MI455X (gfx1250) — matrix operand order repaired and checked by reading, not compiled, not run on hardware
//
#include <hip/hip_runtime.h>
#include <hip/hip_bf16.h>
#include <stdint.h>

// ---------------------------------------------------------------------------
// MI455X / gfx1250 multi-head self-attention, bf16 WMMA path with async
// (ASYNCcnt-tracked) global->LDS tile staging, LDS double buffering, and
// DPP16 row_xmask softmax reductions.
// Shapes: B=4, T=2048, D=512, H=8, Hd=64. Compute-bound (~52 GFLOP vs ~3us
// of HBM time at 23.3 TB/s) -> all matmuls via v_wmma_f32_16x16x32_bf16.
// ---------------------------------------------------------------------------

#define DIMD   512
#define NHEAD  8
#define HDIM   64
#define BATCH  4
#define SEQ    2048
#define MTOT   (BATCH * SEQ)   // 8192 rows for the projection GEMMs

typedef __attribute__((ext_vector_type(16))) __bf16    v16bf;
typedef __attribute__((ext_vector_type(8)))  float     v8f;
typedef __attribute__((ext_vector_type(8)))  uint32_t  v8u;

union Frag16 {
    v16bf    bf;
    v8u      u;
    uint32_t w[8];
};

// Proxy for "this toolchain has the gfx1250 async global->LDS instructions"
// (builtin existence was verified by a round-2 type-check error).
#define HAS_ASYNC_LDS __has_builtin(__builtin_amdgcn_global_load_async_to_lds_b64)

// One 8-byte lane-transfer: global -> LDS. Inline asm (not the builtin) so the
// compiler does not insert conservative loadcnt drains around the barrier --
// ordering is enforced solely by our explicit s_wait_asynccnt.
__device__ __forceinline__ void async_copy_b64(const void* gptr, void* lptr) {
#if HAS_ASYNC_LDS
    uint64_t ga = (uint64_t)gptr;
    uint32_t la = (uint32_t)(uint64_t)lptr;   // generic addr[31:0] == LDS offset
    asm volatile("global_load_async_to_lds_b64 %0, %1, off"
                 :: "v"(la), "v"(ga) : "memory");
#else
    *(unsigned long long*)lptr = *(const unsigned long long*)gptr;
#endif
}

template <int N>
__device__ __forceinline__ void wait_asynccnt() {
#if HAS_ASYNC_LDS
    asm volatile("s_wait_asynccnt %0" ::"i"(N) : "memory");
#endif
}

// Butterfly max across the 16-lane half (both halves reduce independently).
#if __has_builtin(__builtin_amdgcn_update_dpp)
template <int MASK>
__device__ __forceinline__ float xmax_step(float v) {
    // DPP16 row_xmask (0x160|MASK): src lane = lane ^ MASK, pure VALU.
    int x = __builtin_bit_cast(int, v);
    int y = __builtin_amdgcn_update_dpp(x, x, 0x160 | MASK, 0xf, 0xf, true);
    return fmaxf(v, __builtin_bit_cast(float, y));
}
__device__ __forceinline__ float xmax16(float v) {
    v = xmax_step<1>(v);
    v = xmax_step<2>(v);
    v = xmax_step<4>(v);
    v = xmax_step<8>(v);
    return v;
}
#else
__device__ __forceinline__ float xmax16(float v) {
    v = fmaxf(v, __shfl_xor(v, 1));
    v = fmaxf(v, __shfl_xor(v, 2));
    v = fmaxf(v, __shfl_xor(v, 4));
    v = fmaxf(v, __shfl_xor(v, 8));
    return v;
}
#endif

// --------------------------- f32 -> bf16 convert ---------------------------
__global__ void cvt_f32_bf16(const float* __restrict__ src,
                             __bf16* __restrict__ dst, int n) {
    int i = blockIdx.x * blockDim.x + threadIdx.x;
    int stride = gridDim.x * blockDim.x;
    for (; i < n; i += stride) dst[i] = (__bf16)src[i];
}

// ------------------------------- bf16 GEMM ---------------------------------
// C[m,n] = sum_k A[m,k] * W[n,k]   (A:[MTOT,512] bf16, W:[512,512] bf16)
// MODE 0: bf16 out in [B,H,T,Hd]      (Q, K)
// MODE 2: bf16 out in [B,H,Hd,T]      (V transposed -> async-friendly tiles)
// MODE 1: f32  out in [MTOT,512]      (final projection -> d_out)
// WG: 256 threads = 8 waves (2m x 4n); WG tile 128x128; wave tile 64x32
// (4x2 WMMA accumulators). LDS double-buffered, async b64 staging.
template <int MODE>
__global__ __launch_bounds__(256) void gemm_bf16(
    const __bf16* __restrict__ A, const __bf16* __restrict__ W,
    __bf16* __restrict__ out_bf, float* __restrict__ out_f) {
    // pitch 36 elems (72 B): 18-dword bank stride (conflict-free for 16 rows),
    // rows stay 8B-aligned for async b64 writes.
    __shared__ __align__(16) __bf16 lA[2][128 * 36];
    __shared__ __align__(16) __bf16 lW[2][128 * 36];

    const int tid  = threadIdx.x;
    const int lane = tid & 31;
    const int wave = tid >> 5;
    const int wmw  = wave >> 2;     // 0..1 : 64-row group
    const int wn2  = wave & 3;      // 0..3 : 32-col group
    const int lr   = lane & 15;
    const int hh   = lane >> 4;
    const int m0   = blockIdx.y * 128;
    const int n0   = blockIdx.x * 128;

    const unsigned long long* gA8 = (const unsigned long long*)A;  // 4 bf16
    const unsigned long long* gW8 = (const unsigned long long*)W;

    v8f acc[4][2];
#pragma unroll
    for (int mt = 0; mt < 4; ++mt)
#pragma unroll
        for (int nt = 0; nt < 2; ++nt)
            acc[mt][nt] = (v8f){0.f, 0.f, 0.f, 0.f, 0.f, 0.f, 0.f, 0.f};

    // stage tile at k0 into buffer `buf`: per-thread 4 (A) + 4 (W) b64 ops
    auto stage = [&](int k0, int buf) {
#pragma unroll
        for (int i = 0; i < 4; ++i) {               // A: 128 rows x 8 units
            int u = tid + i * 256;
            int row = u >> 3, uc = u & 7;
            async_copy_b64(&gA8[(size_t)(m0 + row) * (DIMD / 4) + (k0 >> 2) + uc],
                           &lA[buf][row * 36 + uc * 4]);
        }
#pragma unroll
        for (int i = 0; i < 4; ++i) {               // W: 128 rows x 8 units
            int u = tid + i * 256;
            int row = u >> 3, uc = u & 7;
            async_copy_b64(&gW8[(size_t)(n0 + row) * (DIMD / 4) + (k0 >> 2) + uc],
                           &lW[buf][row * 36 + uc * 4]);
        }
    };

    stage(0, 0);
    for (int k0 = 0, it = 0; k0 < DIMD; k0 += 32, ++it) {
        const int buf = it & 1;
        if (k0 + 32 < DIMD) {
            stage(k0 + 32, buf ^ 1);   // prefetch next tile into other buffer
            wait_asynccnt<8>();        // current tile's 8 ops have landed
        } else {
            wait_asynccnt<0>();
        }
        __syncthreads();

        Frag16 a[4], b[2];
#pragma unroll
        for (int mt = 0; mt < 4; ++mt) {
            int row = wmw * 64 + mt * 16 + lr;      // A-frag: M = lane&15
#pragma unroll
            for (int j = 0; j < 8; ++j) {
                int k = ((j >> 2) << 4) + (hh << 3) + ((j & 3) << 1);
                a[mt].w[j] = *(const uint32_t*)&lA[buf][row * 36 + k];
            }
        }
#pragma unroll
        for (int nt = 0; nt < 2; ++nt) {
            int row = wn2 * 32 + nt * 16 + lr;      // B-frag: N = lane&15
#pragma unroll
            for (int j = 0; j < 8; ++j) {
                int k = ((j >> 2) << 4) + (hh << 3) + ((j & 3) << 1);
                b[nt].w[j] = *(const uint32_t*)&lW[buf][row * 36 + k];
            }
        }
#pragma unroll
        for (int mt = 0; mt < 4; ++mt)
#pragma unroll
            for (int nt = 0; nt < 2; ++nt)
                acc[mt][nt] = __builtin_amdgcn_wmma_f32_16x16x32_bf16(
                    false, a[mt].bf, false, b[nt].bf, (short)0, acc[mt][nt],
                    false, false);
        __syncthreads();   // everyone done reading buf before it is re-filled
    }

    // epilogue: C layout row = r + 8*hh, col = lane&15
#pragma unroll
    for (int mt = 0; mt < 4; ++mt)
#pragma unroll
        for (int nt = 0; nt < 2; ++nt)
#pragma unroll
            for (int r = 0; r < 8; ++r) {
                int m = m0 + wmw * 64 + mt * 16 + r + (hh << 3);
                int n = n0 + wn2 * 32 + nt * 16 + lr;
                float v = acc[mt][nt][r];
                int bb = m >> 11, t = m & (SEQ - 1);
                int h = n >> 6, hd = n & (HDIM - 1);
                if (MODE == 0) {
                    out_bf[(((size_t)(bb * NHEAD + h) * SEQ) + t) * HDIM + hd] =
                        (__bf16)v;
                } else if (MODE == 2) {
                    out_bf[(((size_t)(bb * NHEAD + h) * HDIM) + hd) * SEQ + t] =
                        (__bf16)v;
                } else {
                    out_f[(size_t)m * DIMD + n] = v;
                }
            }
}

// --------------------------- flash attention -------------------------------
// grid: (T/64, H, B); block: 128 threads = 4 waves; wave handles 16 Q rows.
// K: [B,H,T,Hd]; Vt: [B,H,Hd,T] (transposed at projection time so both tiles
// stage as contiguous async b64 copies). Double-buffered K/Vt tiles.
__global__ __launch_bounds__(128) void attn_kernel(
    const __bf16* __restrict__ Q, const __bf16* __restrict__ K,
    const __bf16* __restrict__ Vt, __bf16* __restrict__ O /* [B,T,D] */) {
    __shared__ __align__(16) __bf16 lK[2][32 * 68];   // [kv][d],  pitch 68
    __shared__ __align__(16) __bf16 lVt[2][64 * 36];  // [hd][kv], pitch 36
    __shared__ __align__(16) __bf16 lP[4 * 16 * 34];  // per-wave P patch

    const int tid  = threadIdx.x;
    const int lane = tid & 31;
    const int wave = tid >> 5;
    const int lr   = lane & 15;
    const int hh   = lane >> 4;
    const int b    = blockIdx.z;
    const int h    = blockIdx.y;
    const int q0   = blockIdx.x * 64 + wave * 16;

    const size_t headoff = (size_t)(b * NHEAD + h) * SEQ * HDIM;
    const uint32_t* gQ = (const uint32_t*)(Q + headoff);
    const unsigned long long* gK8  = (const unsigned long long*)(K + headoff);
    const unsigned long long* gVt8 = (const unsigned long long*)(Vt + headoff);

    // Q fragments: A-layout, d-chunks [0,32) and [32,64)
    Frag16 aq[2];
#pragma unroll
    for (int c = 0; c < 2; ++c)
#pragma unroll
        for (int j = 0; j < 8; ++j) {
            int d = c * 32 + ((j >> 2) << 4) + (hh << 3) + ((j & 3) << 1);
            aq[c].w[j] = gQ[(size_t)(q0 + lr) * (HDIM / 2) + (d >> 1)];
        }

    Frag16 ones;   // B-matrix of 1.0bf16: P x ones -> row sums of P
#pragma unroll
    for (int j = 0; j < 8; ++j) ones.w[j] = 0x3F803F80u;

    v8f o_[4];
#pragma unroll
    for (int n = 0; n < 4; ++n)
        o_[n] = (v8f){0.f, 0.f, 0.f, 0.f, 0.f, 0.f, 0.f, 0.f};
    float mrow[8], lsum[8];
#pragma unroll
    for (int r = 0; r < 8; ++r) { mrow[r] = -3.0e38f; lsum[r] = 0.f; }

    __bf16* myP = lP + wave * 16 * 34;
    const float scale = 0.125f;  // 1/sqrt(64)

    // stage KV tile at kv0: per-thread 4 (K) + 4 (Vt) async b64 ops
    auto stage = [&](int kv0, int buf) {
#pragma unroll
        for (int i = 0; i < 4; ++i) {              // K: 32 rows x 16 units
            int u = tid + i * 128;
            int row = u >> 4, uc = u & 15;
            async_copy_b64(&gK8[(size_t)(kv0 + row) * (HDIM / 4) + uc],
                           &lK[buf][row * 68 + uc * 4]);
        }
#pragma unroll
        for (int i = 0; i < 4; ++i) {              // Vt: 64 rows x 8 units
            int u = tid + i * 128;
            int row = u >> 3, uc = u & 7;
            async_copy_b64(&gVt8[(size_t)row * (SEQ / 4) + (kv0 >> 2) + uc],
                           &lVt[buf][row * 36 + uc * 4]);
        }
    };

    stage(0, 0);
    for (int kv0 = 0, it = 0; kv0 < SEQ; kv0 += 32, ++it) {
        const int buf = it & 1;
        if (kv0 + 32 < SEQ) {
            stage(kv0 + 32, buf ^ 1);
            wait_asynccnt<8>();
        } else {
            wait_asynccnt<0>();
        }
        __syncthreads();

        // S = Q K^T : wave's 16 rows x 32 kv cols (2 tiles of 16) -> 4 WMMAs
        v8f s[2];
#pragma unroll
        for (int nt = 0; nt < 2; ++nt) {
            v8f z = (v8f){0.f, 0.f, 0.f, 0.f, 0.f, 0.f, 0.f, 0.f};
#pragma unroll
            for (int c = 0; c < 2; ++c) {
                Frag16 bk;
#pragma unroll
                for (int j = 0; j < 8; ++j) {
                    int d = c * 32 + ((j >> 2) << 4) + (hh << 3) + ((j & 3) << 1);
                    bk.w[j] =
                        *(const uint32_t*)&lK[buf][(nt * 16 + lr) * 68 + d];
                }
                z = __builtin_amdgcn_wmma_f32_16x16x32_bf16(
                    false, aq[c].bf, false, bk.bf, (short)0, z, false, false);
            }
            s[nt] = z;
        }

        // online softmax: rows live at (r, lane-half); max via DPP row_xmask
#pragma unroll
        for (int r = 0; r < 8; ++r) {
            float a0 = s[0][r] * scale, a1 = s[1][r] * scale;
            float bm    = xmax16(fmaxf(a0, a1));
            float nm    = fmaxf(mrow[r], bm);
            float alpha = __expf(mrow[r] - nm);
            mrow[r]     = nm;
            s[0][r] = __expf(a0 - nm);
            s[1][r] = __expf(a1 - nm);
            lsum[r] *= alpha;
#pragma unroll
            for (int n = 0; n < 4; ++n) o_[n][r] *= alpha;
        }

        // P: C-layout -> LDS -> A-layout (same-wave DS ops are in-order)
#pragma unroll
        for (int r = 0; r < 8; ++r) {
            int row = r + (hh << 3);
            myP[row * 34 + lr]      = (__bf16)s[0][r];
            myP[row * 34 + 16 + lr] = (__bf16)s[1][r];
        }
        Frag16 ap;
#pragma unroll
        for (int j = 0; j < 8; ++j) {
            int k = ((j >> 2) << 4) + (hh << 3) + ((j & 3) << 1);
            ap.w[j] = *(const uint32_t*)&myP[lr * 34 + k];
        }

        // row sums of P via WMMA against ones (instead of lane shuffles)
        v8f ss = __builtin_amdgcn_wmma_f32_16x16x32_bf16(
            false, ap.bf, false, ones.bf, (short)0,
            (v8f){0.f, 0.f, 0.f, 0.f, 0.f, 0.f, 0.f, 0.f}, false, false);
#pragma unroll
        for (int r = 0; r < 8; ++r) lsum[r] += ss[r];

        // O += P V   (4 WMMAs across hd tiles)
#pragma unroll
        for (int n = 0; n < 4; ++n) {
            Frag16 bv;
#pragma unroll
            for (int j = 0; j < 8; ++j) {
                int kv = ((j >> 2) << 4) + (hh << 3) + ((j & 3) << 1);
                bv.w[j] =
                    *(const uint32_t*)&lVt[buf][(n * 16 + lr) * 36 + kv];
            }
            o_[n] = __builtin_amdgcn_wmma_f32_16x16x32_bf16(
                false, ap.bf, false, bv.bf, (short)0, o_[n], false, false);
        }
        __syncthreads();
    }

    // normalize and write [B,T,D] (heads re-merged) as bf16
    __bf16* gO = O + (size_t)b * SEQ * DIMD + h * HDIM;
#pragma unroll
    for (int n = 0; n < 4; ++n)
#pragma unroll
        for (int r = 0; r < 8; ++r) {
            int t  = q0 + r + (hh << 3);
            int hd = n * 16 + lr;
            gO[(size_t)t * DIMD + hd] = (__bf16)(o_[n][r] / lsum[r]);
        }
}

// ------------------------------- launcher ----------------------------------
extern "C" void kernel_launch(void* const* d_in, const int* in_sizes, int n_in,
                              void* d_out, int out_size, void* d_ws,
                              size_t ws_size, hipStream_t stream) {
    const float* x  = (const float*)d_in[0];
    const float* Wq = (const float*)d_in[1];
    const float* Wk = (const float*)d_in[2];
    const float* Wv = (const float*)d_in[3];
    const float* Wo = (const float*)d_in[4];

    const size_t NX = (size_t)MTOT * DIMD;   // 4,194,304
    const size_t NW = (size_t)DIMD * DIMD;   // 262,144

    __bf16* xb  = (__bf16*)d_ws;
    __bf16* wqb = xb + NX;
    __bf16* wkb = wqb + NW;
    __bf16* wvb = wkb + NW;
    __bf16* wob = wvb + NW;
    __bf16* Qb  = wob + NW;   // [B,H,T,Hd]
    __bf16* Kb  = Qb + NX;    // [B,H,T,Hd]
    __bf16* Vtb = Kb + NX;    // [B,H,Hd,T]
    __bf16* Ab  = Vtb + NX;   // attention output, [B,T,D]

    cvt_f32_bf16<<<2048, 256, 0, stream>>>(x, xb, (int)NX);
    cvt_f32_bf16<<<512, 256, 0, stream>>>(Wq, wqb, (int)NW);
    cvt_f32_bf16<<<512, 256, 0, stream>>>(Wk, wkb, (int)NW);
    cvt_f32_bf16<<<512, 256, 0, stream>>>(Wv, wvb, (int)NW);
    cvt_f32_bf16<<<512, 256, 0, stream>>>(Wo, wob, (int)NW);

    dim3 ggrid(DIMD / 128, MTOT / 128);      // (4, 64)
    gemm_bf16<0><<<ggrid, 256, 0, stream>>>(xb, wqb, Qb, nullptr);
    gemm_bf16<0><<<ggrid, 256, 0, stream>>>(xb, wkb, Kb, nullptr);
    gemm_bf16<2><<<ggrid, 256, 0, stream>>>(xb, wvb, Vtb, nullptr);

    attn_kernel<<<dim3(SEQ / 64, NHEAD, BATCH), 128, 0, stream>>>(Qb, Kb, Vtb,
                                                                  Ab);

    gemm_bf16<1><<<ggrid, 256, 0, stream>>>(Ab, wob, nullptr, (float*)d_out);
}
